// BiMambaBlock_13134009991207
// MI455X (gfx1250) — matrix operand order repaired and checked by reading, not compiled, not run on hardware
//
#include <hip/hip_runtime.h>
#include <hip/hip_bf16.h>

// ---------------- types ----------------
typedef __bf16 bf16_t;
typedef __attribute__((ext_vector_type(16))) __bf16 v16bf;
typedef __attribute__((ext_vector_type(8)))  float  v8f;

#define BATCH  4
#define SEQ    2048
#define DMODEL 1024
#define DINNER 2048
#define DSTATE 64
#define DTRANK 64
#define NROWS  ((size_t)BATCH * SEQ)   /* 8192 */

__device__ __forceinline__ bf16_t f2bf(float f) {
  unsigned u = __float_as_uint(f);
  u += 0x7FFFu + ((u >> 16) & 1u);            // round-to-nearest-even
  unsigned short s = (unsigned short)(u >> 16);
  return __builtin_bit_cast(bf16_t, s);
}
__device__ __forceinline__ float bf2f(bf16_t v) {
  unsigned short s = __builtin_bit_cast(unsigned short, v);
  return __uint_as_float(((unsigned)s) << 16);
}

// gfx1250 async global->LDS copy (16B), tracked by ASYNCcnt.
// LDS dest = low 32 bits of the generic shared pointer (ISA 10.2: LDS generic
// addresses carry the wave-relative LDS byte offset in addr[31:0]).
__device__ __forceinline__ void async_copy_b128(void* lds, const void* glob) {
  unsigned loff = (unsigned)(uintptr_t)lds;
  asm volatile("global_load_async_to_lds_b128 %0, %1, off"
               :: "v"(loff), "v"(glob) : "memory");
}
__device__ __forceinline__ void wait_asynccnt0() {
  asm volatile("s_wait_asynccnt 0x0" ::: "memory");
}

// ---------------- WMMA GEMM: C[M,N] = A[M,K] * W[N,K]^T ----------------
// A row-major bf16, W row-major bf16 (einsum 'md,nd->mn'), fp32 accumulate.
// Block tile 128x64, 8 waves (4M x 2N), each wave computes 32x32 via four
// 16x16x32 bf16 WMMAs per K-step. LDS double-buffered, filled with
// global_load_async_to_lds_b128 (ASYNCcnt) overlapping the WMMA of the
// current stage. Requires M%128==0, N%64==0, K%32==0 (true for all shapes).
__global__ __launch_bounds__(256) void gemm_bf16_wmma(
    const bf16_t* __restrict__ A, const bf16_t* __restrict__ W,
    float* __restrict__ C, bf16_t* __restrict__ Cb, int M, int N, int K) {
  // padded rows: 40 bf16 = 80B (16B aligned, spreads LDS banks)
  __shared__ __align__(16) bf16_t sA[2][128][40];
  __shared__ __align__(16) bf16_t sB[2][64][40];
  const int tid  = threadIdx.x;
  const int wave = tid >> 5;
  const int lane = tid & 31;
  const int hi   = lane >> 4;       // half-wave select (ISA 7.12.2)
  const int lr   = lane & 15;
  const int m0 = blockIdx.y * 128;
  const int n0 = blockIdx.x * 64;
  const int wm = (wave & 3) * 32;   // wave M sub-tile (2x16)
  const int wn = (wave >> 2) * 32;  // wave N sub-tile (2x16)
  v8f acc00 = {}; v8f acc01 = {}; v8f acc10 = {}; v8f acc11 = {};

  // cooperative async fill of one stage: A 128x32 (2 chunks/thread) + B 64x32
  auto load_stage = [&](int buf, int ks) {
#pragma unroll
    for (int q = 0; q < 2; ++q) {
      const int id = tid + q * 256;          // 512 x 16B chunks for A
      const int rr = id >> 2;
      const int cc = (id & 3) * 8;
      async_copy_b128(&sA[buf][rr][cc], &A[(size_t)(m0 + rr) * K + ks + cc]);
    }
    const int rr = tid >> 2;                 // 256 x 16B chunks for B
    const int cc = (tid & 3) * 8;
    async_copy_b128(&sB[buf][rr][cc], &W[(size_t)(n0 + rr) * K + ks + cc]);
  };

  const int nk = K >> 5;
  load_stage(0, 0);
  wait_asynccnt0();
  __syncthreads();

  for (int t = 0; t < nk; ++t) {
    const int p = t & 1;
    if (t + 1 < nk) load_stage(p ^ 1, (t + 1) << 5);   // overlap with WMMA

    v16bf a0, a1, b0, b1;
#pragma unroll
    for (int e = 0; e < 16; ++e) {
      // A 16x32 bf16 layout: k = e + 8*((e>>3) + hi)  -> two b128 runs
      const int ka = e + 8 * ((e >> 3) + hi);
      // B 32x16 layout: k = e + 8*((e>>3) + hi)  -> two b128 runs
      const int kb = e + 8 * ((e >> 3) + hi);
      a0[e] = sA[p][wm + lr][ka];
      a1[e] = sA[p][wm + 16 + lr][ka];
      b0[e] = sB[p][wn + lr][kb];
      b1[e] = sB[p][wn + 16 + lr][kb];
    }
    acc00 = __builtin_amdgcn_wmma_f32_16x16x32_bf16(false, a0, false, b0,
                                                    (short)0, acc00, false, false);
    acc01 = __builtin_amdgcn_wmma_f32_16x16x32_bf16(false, a0, false, b1,
                                                    (short)0, acc01, false, false);
    acc10 = __builtin_amdgcn_wmma_f32_16x16x32_bf16(false, a1, false, b0,
                                                    (short)0, acc10, false, false);
    acc11 = __builtin_amdgcn_wmma_f32_16x16x32_bf16(false, a1, false, b1,
                                                    (short)0, acc11, false, false);

    // one barrier per stage: guarantees (a) everyone's async loads for stage
    // t+1 landed, (b) everyone finished reading buffer p before stage t+2
    // overwrites it.
    wait_asynccnt0();
    __syncthreads();
  }

  // C/D layout: lanes 0-15 hold M=g, lanes 16-31 hold M=8+g; N = lane&15
#pragma unroll
  for (int g = 0; g < 8; ++g) {
    const int mr0 = m0 + wm + hi * 8 + g;
    const size_t base0 = (size_t)mr0 * N + n0 + wn + lr;
    const size_t base1 = (size_t)(mr0 + 16) * N + n0 + wn + lr;
    if (C) {
      C[base0] = acc00[g]; C[base0 + 16] = acc01[g];
      C[base1] = acc10[g]; C[base1 + 16] = acc11[g];
    }
    if (Cb) {
      Cb[base0] = f2bf(acc00[g]); Cb[base0 + 16] = f2bf(acc01[g]);
      Cb[base1] = f2bf(acc10[g]); Cb[base1 + 16] = f2bf(acc11[g]);
    }
  }
}

// ---------------- elementwise kernels ----------------
__global__ void cast_bf16_kernel(const float* __restrict__ src,
                                 bf16_t* __restrict__ dst, size_t n) {
  size_t i = (size_t)blockIdx.x * blockDim.x + threadIdx.x;
  if (i < n) dst[i] = f2bf(src[i]);
}

// x -> bf16, plus a sequence-flipped copy for the backward direction
__global__ void cast_x_flip_kernel(const float* __restrict__ x,
                                   bf16_t* __restrict__ xf, bf16_t* __restrict__ xb) {
  size_t i = (size_t)blockIdx.x * blockDim.x + threadIdx.x;
  if (i >= NROWS * DMODEL) return;
  size_t row = i >> 10;             // /DMODEL
  int cc = (int)(i & 1023);
  int b = (int)(row >> 11);         // /SEQ
  int l = (int)(row & 2047);
  bf16_t v = f2bf(x[i]);
  xf[i] = v;
  xb[((size_t)b * SEQ + (SEQ - 1 - l)) * DMODEL + cc] = v;
}

// depthwise conv (width 4, causal pad) + SiLU; reads xs = xz[:, :DINNER]
__global__ void conv_silu_kernel(const bf16_t* __restrict__ xz,
                                 const float* __restrict__ cw, const float* __restrict__ cb,
                                 float* __restrict__ xc, bf16_t* __restrict__ xcb) {
  size_t i = (size_t)blockIdx.x * blockDim.x + threadIdx.x;
  if (i >= NROWS * DINNER) return;
  int d = (int)(i & 2047);
  size_t row = i >> 11;
  int b = (int)(row >> 11);
  int l = (int)(row & 2047);
  float acc = cb[d];
#pragma unroll
  for (int j = 0; j < 4; ++j) {
    int ls = l - 3 + j;
    if (ls >= 0)
      acc += cw[d * 4 + j] * bf2f(xz[((size_t)b * SEQ + ls) * (2 * DINNER) + d]);
  }
  float s = acc / (1.f + __expf(-acc));   // silu
  xc[i] = s;
  xcb[i] = f2bf(s);
}

// dbc[:, :DTRANK] -> bf16 for dt_proj GEMM
__global__ void slice_cast_kernel(const float* __restrict__ dbc,
                                  bf16_t* __restrict__ dtin) {
  size_t i = (size_t)blockIdx.x * blockDim.x + threadIdx.x;
  if (i >= NROWS * DTRANK) return;
  size_t row = i >> 6;
  int rr = (int)(i & 63);
  dtin[i] = f2bf(dbc[row * (DTRANK + 2 * DSTATE) + rr]);
}

// dt = softplus(dt_raw + bias), in place
__global__ void softplus_kernel(float* __restrict__ dt, const float* __restrict__ dtb) {
  size_t i = (size_t)blockIdx.x * blockDim.x + threadIdx.x;
  if (i >= NROWS * DINNER) return;
  int d = (int)(i & 2047);
  float v = dt[i] + dtb[d];
  dt[i] = (v > 20.f) ? v : log1pf(__expf(v));
}

// selective scan, fused gate epilogue: y16 = (scan_y + xc*D) * silu(z)
// one thread per (b,d); h[64] state in VGPRs; B/C rows staged in LDS per step
__global__ __launch_bounds__(64) void scan_kernel(
    const float* __restrict__ dt, const float* __restrict__ xc,
    const float* __restrict__ dbc, const bf16_t* __restrict__ xz,
    const float* __restrict__ alog, const float* __restrict__ Dp,
    bf16_t* __restrict__ y16) {
  __shared__ float sA[64][65];   // +1 pad -> conflict-free per-thread rows
  __shared__ float sBm[64];
  __shared__ float sCm[64];
  const int tid = threadIdx.x;
  const int b    = blockIdx.x >> 5;     // 32 d-blocks per batch
  const int dblk = blockIdx.x & 31;
  const int d = dblk * 64 + tid;
#pragma unroll 8
  for (int n = 0; n < 64; ++n)
    sA[tid][n] = -__expf(alog[(size_t)d * 64 + n]);
  float h[64];
#pragma unroll
  for (int n = 0; n < 64; ++n) h[n] = 0.f;
  const float Dd = Dp[d];
  __syncthreads();

  for (int l = 0; l < SEQ; ++l) {
    size_t row = (size_t)b * SEQ + l;
    sBm[tid] = dbc[row * 192 + 64 + tid];   // B block of dbc
    sCm[tid] = dbc[row * 192 + 128 + tid];  // C block of dbc
    __syncthreads();
    const float dtv = dt[row * DINNER + d];
    const float xv  = xc[row * DINNER + d];
    const float dtx = dtv * xv;
    float y = 0.f;
#pragma unroll
    for (int n = 0; n < 64; ++n) {
      float dA = __expf(dtv * sA[tid][n]);
      h[n] = dA * h[n] + dtx * sBm[n];
      y += h[n] * sCm[n];
    }
    float zv = bf2f(xz[row * (2 * DINNER) + DINNER + d]);
    float yo = (y + xv * Dd) * (zv / (1.f + __expf(-zv)));
    y16[row * DINNER + d] = f2bf(yo);
    __syncthreads();
  }
}

// combined = [out_fwd, flip(out_bwd)] -> bf16
__global__ void combine_kernel(const float* __restrict__ of,
                               const float* __restrict__ ob,
                               bf16_t* __restrict__ comb) {
  size_t i = (size_t)blockIdx.x * blockDim.x + threadIdx.x;
  if (i >= NROWS * 2 * DMODEL) return;
  int e = (int)(i & 2047);
  size_t row = i >> 11;
  int b = (int)(row >> 11);
  int l = (int)(row & 2047);
  float v;
  if (e < DMODEL) v = of[row * DMODEL + e];
  else v = ob[((size_t)b * SEQ + (SEQ - 1 - l)) * DMODEL + (e - DMODEL)];
  comb[i] = f2bf(v);
}

// bias + residual + LayerNorm, one block per row
__global__ __launch_bounds__(256) void ln_kernel(
    const float* __restrict__ outp, const float* __restrict__ pb,
    const float* __restrict__ x, const float* __restrict__ gamma,
    const float* __restrict__ beta, float* __restrict__ out) {
  const int row = blockIdx.x;
  const int tid = threadIdx.x;
  __shared__ float ssum[256];
  __shared__ float ssq[256];
  float hv[4];
  float s = 0.f, sq = 0.f;
#pragma unroll
  for (int i = 0; i < 4; ++i) {
    int cc = tid + i * 256;
    float v = outp[(size_t)row * DMODEL + cc] + pb[cc] + x[(size_t)row * DMODEL + cc];
    hv[i] = v; s += v; sq += v * v;
  }
  ssum[tid] = s; ssq[tid] = sq;
  __syncthreads();
  for (int st = 128; st > 0; st >>= 1) {
    if (tid < st) { ssum[tid] += ssum[tid + st]; ssq[tid] += ssq[tid + st]; }
    __syncthreads();
  }
  float mu = ssum[0] * (1.f / DMODEL);
  float var = ssq[0] * (1.f / DMODEL) - mu * mu;
  float rstd = rsqrtf(var + 1e-5f);
#pragma unroll
  for (int i = 0; i < 4; ++i) {
    int cc = tid + i * 256;
    out[(size_t)row * DMODEL + cc] = (hv[i] - mu) * rstd * gamma[cc] + beta[cc];
  }
}

// ---------------- host orchestration ----------------
extern "C" void kernel_launch(void* const* d_in, const int* in_sizes, int n_in,
                              void* d_out, int out_size, void* d_ws, size_t ws_size,
                              hipStream_t stream) {
  (void)in_sizes; (void)n_in; (void)out_size; (void)ws_size;
  // input order: x, fwd_params{in_proj_w,conv_w,conv_b,x_proj_w,dt_proj_w,
  //   dt_proj_b,A_log,D,out_proj_w}, bwd_params{...}, proj_w, proj_b, gamma, beta
  const float* x      = (const float*)d_in[0];
  const float* proj_w = (const float*)d_in[19];
  const float* proj_b = (const float*)d_in[20];
  const float* gamma  = (const float*)d_in[21];
  const float* beta   = (const float*)d_in[22];

  char* wsp = (char*)d_ws;
  size_t off = 0;
  auto alloc = [&](size_t bytes) -> char* {
    char* rp = wsp + off;
    off += (bytes + 255) & ~(size_t)255;
    return rp;
  };

  // persistent buffers
  bf16_t* xf16 = (bf16_t*)alloc(NROWS * DMODEL * 2);
  bf16_t* xb16 = (bf16_t*)alloc(NROWS * DMODEL * 2);
  bf16_t *ipw16[2], *xpw16[2], *dtw16[2], *opw16[2];
  for (int dir = 0; dir < 2; ++dir) {
    ipw16[dir] = (bf16_t*)alloc((size_t)2 * DINNER * DMODEL * 2);
    xpw16[dir] = (bf16_t*)alloc((size_t)192 * DINNER * 2);
    dtw16[dir] = (bf16_t*)alloc((size_t)DINNER * DTRANK * 2);
    opw16[dir] = (bf16_t*)alloc((size_t)DMODEL * DINNER * 2);
  }
  bf16_t* pw16 = (bf16_t*)alloc((size_t)DMODEL * 2 * DMODEL * 2);
  float* out_dir0 = (float*)alloc(NROWS * DMODEL * 4);
  float* out_dir1 = (float*)alloc(NROWS * DMODEL * 4);
  float* out_dirs[2] = {out_dir0, out_dir1};
  bf16_t* comb16 = (bf16_t*)alloc(NROWS * 2 * DMODEL * 2);
  float* outp = (float*)alloc(NROWS * DMODEL * 4);
  // per-direction scratch (reused across the two passes)
  bf16_t* xz16   = (bf16_t*)alloc(NROWS * 2 * DINNER * 2);
  float*  xcf    = (float*)alloc(NROWS * DINNER * 4);
  bf16_t* xc16   = (bf16_t*)alloc(NROWS * DINNER * 2);
  float*  dbc    = (float*)alloc(NROWS * 192 * 4);
  bf16_t* dtin16 = (bf16_t*)alloc(NROWS * DTRANK * 2);
  float*  dtbuf  = (float*)alloc(NROWS * DINNER * 4);
  bf16_t* y16    = (bf16_t*)alloc(NROWS * DINNER * 2);

  auto nblocks = [](size_t n) { return (unsigned)((n + 255) / 256); };

  cast_x_flip_kernel<<<nblocks(NROWS * DMODEL), 256, 0, stream>>>(x, xf16, xb16);
  for (int dir = 0; dir < 2; ++dir) {
    const float* ipw = (const float*)d_in[1 + dir * 9 + 0];
    const float* xpw = (const float*)d_in[1 + dir * 9 + 3];
    const float* dtw = (const float*)d_in[1 + dir * 9 + 4];
    const float* opw = (const float*)d_in[1 + dir * 9 + 8];
    cast_bf16_kernel<<<nblocks((size_t)2 * DINNER * DMODEL), 256, 0, stream>>>(
        ipw, ipw16[dir], (size_t)2 * DINNER * DMODEL);
    cast_bf16_kernel<<<nblocks((size_t)192 * DINNER), 256, 0, stream>>>(
        xpw, xpw16[dir], (size_t)192 * DINNER);
    cast_bf16_kernel<<<nblocks((size_t)DINNER * DTRANK), 256, 0, stream>>>(
        dtw, dtw16[dir], (size_t)DINNER * DTRANK);
    cast_bf16_kernel<<<nblocks((size_t)DMODEL * DINNER), 256, 0, stream>>>(
        opw, opw16[dir], (size_t)DMODEL * DINNER);
  }
  cast_bf16_kernel<<<nblocks((size_t)DMODEL * 2 * DMODEL), 256, 0, stream>>>(
      proj_w, pw16, (size_t)DMODEL * 2 * DMODEL);

  for (int dir = 0; dir < 2; ++dir) {
    const float* cw   = (const float*)d_in[1 + dir * 9 + 1];
    const float* cb   = (const float*)d_in[1 + dir * 9 + 2];
    const float* dtb  = (const float*)d_in[1 + dir * 9 + 5];
    const float* alog = (const float*)d_in[1 + dir * 9 + 6];
    const float* Dp   = (const float*)d_in[1 + dir * 9 + 7];
    const bf16_t* xin = dir ? xb16 : xf16;

    // xz = x @ in_proj_w^T  (8192x4096, K=1024) -> bf16
    gemm_bf16_wmma<<<dim3(4096 / 64, 8192 / 128), 256, 0, stream>>>(
        xin, ipw16[dir], nullptr, xz16, 8192, 4096, 1024);
    conv_silu_kernel<<<nblocks(NROWS * DINNER), 256, 0, stream>>>(
        xz16, cw, cb, xcf, xc16);
    // dbc = xc @ x_proj_w^T (8192x192, K=2048) -> fp32
    gemm_bf16_wmma<<<dim3(192 / 64, 8192 / 128), 256, 0, stream>>>(
        xc16, xpw16[dir], dbc, nullptr, 8192, 192, 2048);
    slice_cast_kernel<<<nblocks(NROWS * DTRANK), 256, 0, stream>>>(dbc, dtin16);
    // dt_raw = dt_in @ dt_proj_w^T (8192x2048, K=64) -> fp32
    gemm_bf16_wmma<<<dim3(2048 / 64, 8192 / 128), 256, 0, stream>>>(
        dtin16, dtw16[dir], dtbuf, nullptr, 8192, 2048, 64);
    softplus_kernel<<<nblocks(NROWS * DINNER), 256, 0, stream>>>(dtbuf, dtb);
    // selective scan + gate -> y16 (bf16)
    scan_kernel<<<BATCH * (DINNER / 64), 64, 0, stream>>>(
        dtbuf, xcf, dbc, xz16, alog, Dp, y16);
    // out_dir = y @ out_proj_w^T (8192x1024, K=2048)
    gemm_bf16_wmma<<<dim3(1024 / 64, 8192 / 128), 256, 0, stream>>>(
        y16, opw16[dir], out_dirs[dir], nullptr, 8192, 1024, 2048);
  }

  combine_kernel<<<nblocks(NROWS * 2 * DMODEL), 256, 0, stream>>>(
      out_dirs[0], out_dirs[1], comb16);
  // out = combined @ proj_w^T (8192x1024, K=2048)
  gemm_bf16_wmma<<<dim3(1024 / 64, 8192 / 128), 256, 0, stream>>>(
      comb16, pw16, outp, nullptr, 8192, 1024, 2048);
  ln_kernel<<<(unsigned)NROWS, 256, 0, stream>>>(
      outp, proj_b, x, gamma, beta, (float*)d_out);
}
